// MaskedConv2D_36644660970101
// MI455X (gfx1250) — matrix operand order repaired and checked by reading, not compiled, not run on hardware
//
#include <hip/hip_runtime.h>

typedef _Float16 f16;
typedef __attribute__((ext_vector_type(8)))  _Float16 v8h;
typedef __attribute__((ext_vector_type(16))) _Float16 v16h;
typedef __attribute__((ext_vector_type(8)))  float    v8f;

#define CIN   64
#define COUT  64
#define Hdim  256
#define Wdim  256
#define HWSZ  (Hdim * Wdim)
#define NWF   (9 * 4 * 64 * 16) /* 36864 f16 per weight array (hi / lo) */
#define XCOLS 130               /* 128 pixels + 2 halo columns          */
#define XSTR  72                /* halves per column: 64 ch + 8 pad     */
#define XROWS 4                 /* 2 output rows + 2 halo rows          */
#define NXF   (XROWS * XCOLS * XSTR) /* 37440 halves per x array        */

// ---------------------------------------------------------------------------
// Repack + split weight [COUT][CIN][3][3] fp32 -> two f16 arrays (hi, lo) in
// B-fragment layout:  w*[((kpos*4 + g)*64 + cout)*16 + e],  cin = g*16 + e.
// A lane's V_WMMA_F32_16X16X32_F16 B fragment is then 32 contiguous bytes.
// ---------------------------------------------------------------------------
__global__ __launch_bounds__(256) void repack_w(const float* __restrict__ w,
                                                f16* __restrict__ whi,
                                                f16* __restrict__ wlo) {
  int i = blockIdx.x * 256 + threadIdx.x;
  if (i >= NWF) return;
  int e    = i & 15;
  int co   = (i >> 4) & 63;
  int kg   = i >> 10;         // kpos*4 + g
  int g    = kg & 3;
  int kpos = kg >> 2;
  int ci   = g * 16 + e;
  float v  = w[(co * CIN + ci) * 9 + kpos];
  f16 hi = (f16)v;
  f16 lo = (f16)(v - (float)hi);   // residual; f16 subnormals preserved by WMMA
  whi[i] = hi;
  wlo[i] = lo;
}

// ---------------------------------------------------------------------------
// Implicit-GEMM masked conv, f16x3 emulated-fp32 on the XDL pipe.
// Block = 2 rows x 128 pixels (8 waves); wave = 32 pixels x 64 couts
// (8 fp32 16x16 accumulators). K = CIN*9 = 576, 32/chunk with
// V_WMMA_F32_16X16X32_F16, 3 passes (hh, lh, hl), fp32 accumulate.
// Both operands staged split in LDS (297 KB of the 320 KB WGP LDS):
//   - weight 147 KB: B fragment = 32B contiguous (2x ds_load_b128)
//   - x tile 146 KB: split to (hi,lo) f16 ONCE per block; A fragment = two
//     16B channel groups (2x ds_load_b128), conflict-free via 72-half stride.
// Inner loop per chunk: 24 ds_load_b128 + 24 v_wmma  (1.0 ds per wmma).
// ---------------------------------------------------------------------------
__global__ __launch_bounds__(256) void mconv(const float* __restrict__ x,
                                             const float* __restrict__ mask,
                                             const float* __restrict__ wsp,
                                             const float* __restrict__ bias,
                                             float* __restrict__ out) {
  __shared__ f16 wlds[2 * NWF];   // 147456 B : weight hi | lo
  __shared__ f16 xhls[NXF];       //  74880 B : x tile hi
  __shared__ f16 xlls[NXF];       //  74880 B : x tile lo

  const int tid  = threadIdx.x;
  const int wave = tid >> 5;
  const int lane = tid & 31;
  const int lm   = lane & 15;   // M (pixel) for A, N (cout) for B/C
  const int lh   = lane >> 4;   // lane-half selector

  // block coordinates (uniform): batch, row pair, 128-pixel half-row
  const int b     = blockIdx.x >> 8;
  const int rem   = blockIdx.x & 255;
  const int h0    = (rem >> 1) << 1;     // first of 2 output rows
  const int wbase = (rem & 1) << 7;      // first of 128 pixels
  // wave coordinates
  const int wrow    = wave >> 2;         // 0/1: which output row
  const int wcol    = wave & 3;          // which 32-pixel group
  const int row_out = h0 + wrow;
  const int w0      = wbase + wcol * 32; // wave's first pixel

  // ---- stage split weight: contiguous float4 copy, fully coalesced
  {
    const float4* src = (const float4*)wsp;
    float4*       dst = (float4*)wlds;
    #pragma unroll
    for (int i = 0; i < (2 * NWF * 2) / 16 / 256; ++i)   // 36 iterations
      dst[tid + i * 256] = src[tid + i * 256];
  }

  // ---- stage split x halo tile (4 rows x 130 cols x 64 ch), convert once
  {
    const float* xg = x + (size_t)b * (CIN * HWSZ);
    for (int i = tid; i < XROWS * CIN * XCOLS; i += 256) {  // 130 iterations
      int col  = i % XCOLS;
      int t    = i / XCOLS;
      int ch   = t & 63;
      int r    = t >> 6;                 // 0..3 row slot (grow = h0 + r - 1)
      int grow = h0 + r - 1;
      int gcol = wbase + col - 1;
      float v = 0.0f;
      if ((unsigned)grow < (unsigned)Hdim && (unsigned)gcol < (unsigned)Wdim)
        v = xg[(size_t)ch * HWSZ + grow * Wdim + gcol];
      f16 hi = (f16)v;
      f16 lo = (f16)(v - (float)hi);
      int o = (r * XCOLS + col) * XSTR + ch;
      xhls[o] = hi;
      xlls[o] = lo;
    }
  }
  __syncthreads();

  v8f c[8] = {};    // [s*4+n]: M-subtile s (16 px), cout group n

  const v16h* bhiA = (const v16h*)wlds;          // 2304 B fragments (hi)
  const v16h* bloA = (const v16h*)(wlds + NWF);  // 2304 B fragments (lo)

  for (int kh = 0; kh < 3; ++kh) {
    int row = row_out + kh - 1;
    if ((unsigned)row >= (unsigned)Hdim) continue;   // uniform: zero-pad row
    int rslot = wrow + kh;                            // staged row slot 0..3
    #pragma unroll
    for (int kw = 0; kw < 3; ++kw) {
      int kpos = kh * 3 + kw;
      #pragma unroll
      for (int kk = 0; kk < 2; ++kk) {               // 32-channel K chunks
        // ---- A fragments for both 16-pixel sub-tiles
        v16h ah[2], al[2];
        #pragma unroll
        for (int s = 0; s < 2; ++s) {
          int abase = (rslot * XCOLS + (wcol * 32 + s * 16 + lm + kw)) * XSTR
                      + 16 * lh + 32 * kk;
          v8h ph0 = *(const v8h*)(xhls + abase);
          v8h ph1 = *(const v8h*)(xhls + abase + 8);  // +8 channels
          v8h pl0 = *(const v8h*)(xlls + abase);
          v8h pl1 = *(const v8h*)(xlls + abase + 8);
          ah[s] = __builtin_shufflevector(ph0, ph1, 0,1,2,3,4,5,6,7,8,9,10,11,12,13,14,15);
          al[s] = __builtin_shufflevector(pl0, pl1, 0,1,2,3,4,5,6,7,8,9,10,11,12,13,14,15);
        }
        int fb = (kpos * 4 + 2 * kk + lh) * 64 + lm;  // B fragment idx (+n*16)
        #pragma unroll
        for (int n = 0; n < 4; ++n) {
          v16h bh = bhiA[fb + n * 16];
          v16h bl = bloA[fb + n * 16];
          #pragma unroll
          for (int s = 0; s < 2; ++s) {
            c[s*4+n] = __builtin_amdgcn_wmma_f32_16x16x32_f16(false, ah[s], false, bh, (short)0, c[s*4+n], false, false);
            c[s*4+n] = __builtin_amdgcn_wmma_f32_16x16x32_f16(false, al[s], false, bh, (short)0, c[s*4+n], false, false);
            c[s*4+n] = __builtin_amdgcn_wmma_f32_16x16x32_f16(false, ah[s], false, bl, (short)0, c[s*4+n], false, false);
          }
        }
      }
    }
  }

  // ---- validity: one bit per pixel of this wave's 32-pixel tile
  const float* mb   = mask + (size_t)b * HWSZ;
  bool         flag = false;
  #pragma unroll
  for (int dh = -1; dh <= 1; ++dh) {
    int r = row_out + dh;
    if ((unsigned)r >= (unsigned)Hdim) continue;     // uniform
    #pragma unroll
    for (int dw = -1; dw <= 1; ++dw) {
      int   cwp = w0 + lane + dw;                    // pixel index = lane
      bool  ib  = (unsigned)cwp < (unsigned)Wdim;
      int   cwc = ib ? cwp : 0;
      float mv  = mb[r * Wdim + cwc];
      flag = flag || (ib && (mv != 0.0f));
    }
  }
#if __has_builtin(__builtin_amdgcn_ballot_w32)
  unsigned vmask = __builtin_amdgcn_ballot_w32(flag);  // bit p = pixel w0+p
#else
  unsigned vmask = (unsigned)__ballot(flag);
#endif

  // ---- store: lane (s,n,r) -> out[b][n*16+lm][row_out][w0 + s*16 + r + 8*lh]
  // For a fixed cout, r=0..7 over both lane halves fills one whole 64B line.
  #pragma unroll
  for (int s = 0; s < 2; ++s) {
    #pragma unroll
    for (int n = 0; n < 4; ++n) {
      float  bn = bias[n * 16 + lm];
      float* on = out + ((size_t)b * COUT + n * 16 + lm) * HWSZ
                      + (size_t)row_out * Wdim + w0 + s * 16 + 8 * lh;
      #pragma unroll
      for (int r = 0; r < 8; ++r) {
        int m = s * 16 + r + 8 * lh;
        on[r] = ((vmask >> m) & 1u) ? (c[s*4+n][r] + bn) : 0.0f;
      }
    }
  }
}

// ---------------------------------------------------------------------------
extern "C" void kernel_launch(void* const* d_in, const int* in_sizes, int n_in,
                              void* d_out, int out_size, void* d_ws, size_t ws_size,
                              hipStream_t stream) {
  const float* x    = (const float*)d_in[0];
  const float* mask = (const float*)d_in[1];
  const float* w    = (const float*)d_in[2];
  const float* bias = (const float*)d_in[3];
  float*       out  = (float*)d_out;
  f16*         whi  = (f16*)d_ws;          // 73728 B
  f16*         wlo  = whi + NWF;           // 73728 B (total 147456 B scratch)

  repack_w<<<(NWF + 255) / 256, 256, 0, stream>>>(w, whi, wlo);
  mconv<<<2048, 256, 0, stream>>>(x, mask, (const float*)d_ws, bias, out);
}
